// MultiHeadAttention_28484223107494
// MI455X (gfx1250) — matrix operand order repaired and checked by reading, not compiled, not run on hardware
//
#include <hip/hip_runtime.h>
#include <hip/hip_bf16.h>

// ---------------------------------------------------------------------------
// MI455X (gfx1250) multi-head attention, bf16 WMMA with f32 accumulate.
// - All matmuls: v_wmma_f32_16x16x32_bf16 (compute-bound workload, ~103 GFLOP)
// - Flash-attention streaming (no 512MB score tensor)
// - K/V tiles staged into LDS by the Tensor Data Mover (tensor_load_to_lds),
//   double-buffered, shared by all 4 waves of the block (4x traffic cut)
// - GEMM K-loops software-pipelined so loads overlap WMMAs
// wave32; block = 128 threads = 4 waves.
// ---------------------------------------------------------------------------

typedef __attribute__((ext_vector_type(16))) __bf16        v16bf;
typedef __attribute__((ext_vector_type(8)))  float         v8f;
typedef __attribute__((ext_vector_type(4)))  unsigned int  u32x4;
typedef __attribute__((ext_vector_type(8)))  unsigned int  u32x8;

#define DEV __device__ __forceinline__

constexpr int NH = 16, DM = 1024, DK = 64, BB = 8, SS = 1024;
constexpr int MTOT = BB * SS;  // 8192 rows

DEV unsigned short f2bf(float f) {
  unsigned u = __builtin_bit_cast(unsigned, f);
  u += 0x7FFFu + ((u >> 16) & 1u);          // round-to-nearest-even
  return (unsigned short)(u >> 16);
}

union V16 {
  v16bf v;
  u32x4 q[2];
};

// A-operand tile load (16x32 bf16). p = rowBase + kchunk + kb, kb=(lane>>4)*8.
DEV v16bf loadA(const unsigned short* p) {
  V16 t;
  t.q[0] = *reinterpret_cast<const u32x4*>(p);
  t.q[1] = *reinterpret_cast<const u32x4*>(p + 16);
  return t.v;
}

// B-operand tile load (32x16 bf16). Per-lane elements K = hi*8 + j, 16 + hi*8 + j.
DEV v16bf loadB(const unsigned short* p, int hi) {
  p += hi * 8;
  V16 t;
  t.q[0] = *reinterpret_cast<const u32x4*>(p);
  t.q[1] = *reinterpret_cast<const u32x4*>(p + 16);
  return t.v;
}

DEV v8f wmma_bf16(v16bf a, v16bf b, v8f c) {
  return __builtin_amdgcn_wmma_f32_16x16x32_bf16(
      false, a, false, b, (short)0, c, false, false);
}

DEV v8f zero8() {
  v8f z;
#pragma unroll
  for (int i = 0; i < 8; ++i) z[i] = 0.f;
  return z;
}

DEV float rmax16(float v) {
  v = fmaxf(v, __shfl_xor(v, 1, 32));
  v = fmaxf(v, __shfl_xor(v, 2, 32));
  v = fmaxf(v, __shfl_xor(v, 4, 32));
  v = fmaxf(v, __shfl_xor(v, 8, 32));
  return v;
}
DEV float rsum16(float v) {
  v += __shfl_xor(v, 1, 32);
  v += __shfl_xor(v, 2, 32);
  v += __shfl_xor(v, 4, 32);
  v += __shfl_xor(v, 8, 32);
  return v;
}

// ---------------------------------------------------------------------------
// Tensor Data Mover: 2D tile (elements are 2 bytes) global -> LDS.
// Builds the D# in SGPRs (all inputs wave-uniform) and issues the DMA.
// group0: [1:0]=count=1, [63:32]=lds_addr, [120:64]=global byte addr,
//         [127:126]=type=2.  group1: data_size=2B, dims/strides in elements.
// ---------------------------------------------------------------------------
DEV void tdm_load_2d(unsigned lds_off, const unsigned short* g,
                     unsigned tile0, unsigned tile1,
                     unsigned td0, unsigned td1, unsigned stride0) {
  const unsigned long long ga = (unsigned long long)(uintptr_t)g;
  u32x4 g0;
  g0[0] = 1u;                                         // count=1 (valid, user)
  g0[1] = lds_off;                                    // LDS byte address
  g0[2] = (unsigned)ga;                               // global addr [31:0]
  g0[3] = (unsigned)((ga >> 32) & 0x01FFFFFFu) | 0x80000000u;  // [56:32]|type=2
  u32x8 g1;
  g1[0] = 0x00010000u;                                // data_size = 1 (2 bytes)
  g1[1] = (td0 & 0xFFFFu) << 16;                      // tensor_dim0 [15:0]
  g1[2] = (td0 >> 16) | ((td1 & 0xFFFFu) << 16);      // td0 hi | tensor_dim1 lo
  g1[3] = (td1 >> 16) | (tile0 << 16);                // td1 hi | tile_dim0
  g1[4] = tile1;                                      // tile_dim1 (tile_dim2=0)
  g1[5] = stride0;                                    // tensor_dim0_stride lo
  g1[6] = 0u;
  g1[7] = 0u;
  asm volatile("tensor_load_to_lds %0, %1" :: "s"(g0), "s"(g1) : "memory");
}

// ---------------------------------------------------------------------------
// prep kernels
// ---------------------------------------------------------------------------
__global__ __launch_bounds__(256) void k_cvt(const float* __restrict__ s,
                                             unsigned short* __restrict__ d,
                                             int n) {
  for (int i = blockIdx.x * blockDim.x + threadIdx.x; i < n;
       i += gridDim.x * blockDim.x)
    d[i] = f2bf(s[i]);
}

// w [16][1024][64] -> wt [16][64][1024]
__global__ __launch_bounds__(256) void k_cvt_wT(const float* __restrict__ w,
                                                unsigned short* __restrict__ wt) {
  const int total = NH * DK * DM;
  for (int i = blockIdx.x * blockDim.x + threadIdx.x; i < total;
       i += gridDim.x * blockDim.x) {
    int d = i & (DM - 1);
    int n = (i >> 10) & (DK - 1);
    int h = i >> 16;
    wt[i] = f2bf(w[(h << 16) + (d << 6) + n]);
  }
}

// ---------------------------------------------------------------------------
// QKV projection GEMM, software-pipelined K loop (16x64 tile per wave)
// ---------------------------------------------------------------------------
__global__ __launch_bounds__(128) void qkv_gemm(
    const unsigned short* __restrict__ A,    // qbf [8192][1024]
    const unsigned short* __restrict__ WT,   // [16][64][1024]
    unsigned short* __restrict__ dst, int transposedV) {
  const int lane = threadIdx.x & 31, wv = threadIdx.x >> 5;
  const int r = lane & 15, hi = lane >> 4, kb = hi * 8;
  const int mt = blockIdx.x * 4 + wv;
  const int h = blockIdx.y;
  const unsigned short* aRow = A + (size_t)(mt * 16 + r) * DM;
  const unsigned short* b0 = WT + ((size_t)h * DK + r) * DM;
  v8f acc[4];
#pragma unroll
  for (int t = 0; t < 4; ++t) acc[t] = zero8();

  // pipeline prologue
  v16bf a = loadA(aRow + kb);
  v16bf bt0 = loadB(b0, hi);
  v16bf bt1 = loadB(b0 + 16 * DM, hi);
  v16bf bt2 = loadB(b0 + 32 * DM, hi);
  v16bf bt3 = loadB(b0 + 48 * DM, hi);

  for (int kc = 32; kc < DM; kc += 32) {
    __builtin_prefetch(aRow + kc + 128, 0, 0);
    v16bf an  = loadA(aRow + kc + kb);
    v16bf bn0 = loadB(b0 + kc, hi);
    v16bf bn1 = loadB(b0 + 16 * DM + kc, hi);
    v16bf bn2 = loadB(b0 + 32 * DM + kc, hi);
    v16bf bn3 = loadB(b0 + 48 * DM + kc, hi);
    acc[0] = wmma_bf16(a, bt0, acc[0]);
    acc[1] = wmma_bf16(a, bt1, acc[1]);
    acc[2] = wmma_bf16(a, bt2, acc[2]);
    acc[3] = wmma_bf16(a, bt3, acc[3]);
    a = an; bt0 = bn0; bt1 = bn1; bt2 = bn2; bt3 = bn3;
  }
  acc[0] = wmma_bf16(a, bt0, acc[0]);
  acc[1] = wmma_bf16(a, bt1, acc[1]);
  acc[2] = wmma_bf16(a, bt2, acc[2]);
  acc[3] = wmma_bf16(a, bt3, acc[3]);

#pragma unroll
  for (int t = 0; t < 4; ++t) {
#pragma unroll
    for (int i = 0; i < 8; ++i) {
      const int row = mt * 16 + i + 8 * hi;  // flat m = b*S + s
      const int n = t * 16 + r;
      const unsigned short v = f2bf(acc[t][i]);
      if (!transposedV) {
        dst[((size_t)h * MTOT + row) * DK + n] = v;
      } else {
        const int bb = row >> 10, s = row & (SS - 1);
        dst[(((size_t)h * BB + bb) * DK + n) * SS + s] = v;
      }
    }
  }
}

// ---------------------------------------------------------------------------
// Flash attention. 4 waves/block share (h,b); K/V chunks staged into LDS by
// the TDM (wave 0 issues descriptors), double-buffered to overlap DMA with
// WMMA compute. Each wave owns one 16-query tile.
// ---------------------------------------------------------------------------
__global__ __launch_bounds__(128) void attn_fwd(
    const unsigned short* __restrict__ Qh,   // [h][b*S][64]
    const unsigned short* __restrict__ Kh,   // [h][b*S][64]
    const unsigned short* __restrict__ VT,   // [h][b][64][S]
    const unsigned char* __restrict__ mask,  // [b][S] bool
    unsigned short* __restrict__ attnout)    // [b][S][h*64] bf16
{
  __shared__ unsigned short ldsK[2][32 * 64];   // [buf][key][d]     4KB each
  __shared__ unsigned short ldsV[2][64 * 32];   // [buf][dv][key]    4KB each
  __shared__ unsigned short ldsP[4][16 * 32];   // per-wave P tile   1KB each

  const int lane = threadIdx.x & 31, wv = threadIdx.x >> 5;
  const int r = lane & 15, hi = lane >> 4, kb = hi * 8;
  const int qt = blockIdx.x * 4 + wv;
  const int h = blockIdx.y, b = blockIdx.z;
  const size_t hb = (size_t)h * BB + b;
  const float NEGINF = -__builtin_inff();

  const unsigned short* Kbase = Kh + hb * SS * DK;        // [key][64]
  const unsigned short* Vbase = VT + hb * (size_t)DK * SS; // [dv][S]

  const unsigned short* Qb = Qh + (hb * SS + qt * 16 + r) * DK;
  const v16bf aQ0 = loadA(Qb + kb);
  const v16bf aQ1 = loadA(Qb + 32 + kb);

  float M[8], L[8];
  v8f o[4];
#pragma unroll
  for (int i = 0; i < 8; ++i) { M[i] = NEGINF; L[i] = 0.f; }
#pragma unroll
  for (int t = 0; t < 4; ++t) o[t] = zero8();

  unsigned short* P = ldsP[wv];
  const unsigned char* mrow = mask + b * SS;

  // prologue: DMA chunk 0 into buffer 0
  if (wv == 0) {
    tdm_load_2d((unsigned)(uintptr_t)&ldsK[0][0], Kbase, 64, 32, 64, 32, 64);
    tdm_load_2d((unsigned)(uintptr_t)&ldsV[0][0], Vbase, 32, 64, 32, 64, SS);
    __builtin_amdgcn_s_wait_tensorcnt(0);
  }
  __syncthreads();

  const int nchunks = SS / 32;
  for (int c = 0; c < nchunks; ++c) {
    const int cur = c & 1, nxt = cur ^ 1;
    const int kc = c * 32;
    if (wv == 0 && c + 1 < nchunks) {   // DMA next chunk while computing
      tdm_load_2d((unsigned)(uintptr_t)&ldsK[nxt][0],
                  Kbase + (size_t)(kc + 32) * DK, 64, 32, 64, 32, 64);
      tdm_load_2d((unsigned)(uintptr_t)&ldsV[nxt][0],
                  Vbase + (kc + 32), 32, 64, 32, 64, SS);
    }
    const unsigned short* kbuf = ldsK[cur];
    const unsigned short* vbuf = ldsV[cur];

    // ---- scores: two 16x16 tiles from LDS K ----
    v8f s0 = zero8(), s1 = zero8();
    s0 = wmma_bf16(aQ0, loadB(kbuf + r * 64, hi), s0);
    s0 = wmma_bf16(aQ1, loadB(kbuf + r * 64 + 32, hi), s0);
    s1 = wmma_bf16(aQ0, loadB(kbuf + (16 + r) * 64, hi), s1);
    s1 = wmma_bf16(aQ1, loadB(kbuf + (16 + r) * 64 + 32, hi), s1);

    const float m0 = mrow[kc + r] ? NEGINF : 0.f;
    const float m1 = mrow[kc + 16 + r] ? NEGINF : 0.f;

    // ---- joint online softmax over the 32-key block ----
#pragma unroll
    for (int i = 0; i < 8; ++i) {
      const float sv0 = s0[i] * 0.125f + m0;
      const float sv1 = s1[i] * 0.125f + m1;
      const float tm = rmax16(fmaxf(sv0, sv1));
      const float nm = fmaxf(M[i], tm);
      const bool dead = (nm == NEGINF);
      const float al = dead ? 1.f : __expf(M[i] - nm);
      const float p0 = dead ? 0.f : __expf(sv0 - nm);
      const float p1 = dead ? 0.f : __expf(sv1 - nm);
      const float ts = rsum16(p0 + p1);
      L[i] = L[i] * al + ts;
      M[i] = nm;
      o[0][i] *= al; o[1][i] *= al; o[2][i] *= al; o[3][i] *= al;
      const int prow = (i + 8 * hi) * 32;
      P[prow + r]      = f2bf(p0);
      P[prow + 16 + r] = f2bf(p1);
    }

    // ---- P (16x32) as A operand; accumulate P*V from LDS V ----
    v16bf aP = loadA(P + r * 32 + kb);
    o[0] = wmma_bf16(aP, loadB(vbuf + r * 32, hi), o[0]);
    o[1] = wmma_bf16(aP, loadB(vbuf + (16 + r) * 32, hi), o[1]);
    o[2] = wmma_bf16(aP, loadB(vbuf + (32 + r) * 32, hi), o[2]);
    o[3] = wmma_bf16(aP, loadB(vbuf + (48 + r) * 32, hi), o[3]);

    if (wv == 0) __builtin_amdgcn_s_wait_tensorcnt(0);
    __syncthreads();   // next-chunk DMA complete & this chunk's reads done
  }

  // ---- normalize and store concat-head output [b][s][h*64+dv] ----
#pragma unroll
  for (int i = 0; i < 8; ++i) {
    const float inv = (L[i] > 0.f) ? 1.f / L[i] : 0.f;
    const int row = qt * 16 + i + 8 * hi;
    unsigned short* dst = attnout + ((size_t)b * SS + row) * DM + h * DK;
    dst[r]      = f2bf(o[0][i] * inv);
    dst[16 + r] = f2bf(o[1][i] * inv);
    dst[32 + r] = f2bf(o[2][i] * inv);
    dst[48 + r] = f2bf(o[3][i] * inv);
  }
}

// ---------------------------------------------------------------------------
// Output projection, software-pipelined
// ---------------------------------------------------------------------------
__global__ __launch_bounds__(128) void out_proj(
    const unsigned short* __restrict__ A,    // attn [8192][1024] bf16
    const unsigned short* __restrict__ PW,   // [1024][1024] bf16
    const float* __restrict__ bias,
    float* __restrict__ out) {
  const int lane = threadIdx.x & 31, wv = threadIdx.x >> 5;
  const int r = lane & 15, hi = lane >> 4, kb = hi * 8;
  const int mt = blockIdx.x * 4 + wv;
  const int nt = blockIdx.y;
  const unsigned short* aRow = A + (size_t)(mt * 16 + r) * DM;
  const unsigned short* b0 = PW + ((size_t)nt * 64 + r) * DM;
  v8f acc[4];
#pragma unroll
  for (int t = 0; t < 4; ++t) acc[t] = zero8();

  v16bf a = loadA(aRow + kb);
  v16bf bt0 = loadB(b0, hi);
  v16bf bt1 = loadB(b0 + 16 * DM, hi);
  v16bf bt2 = loadB(b0 + 32 * DM, hi);
  v16bf bt3 = loadB(b0 + 48 * DM, hi);

  for (int kc = 32; kc < DM; kc += 32) {
    __builtin_prefetch(aRow + kc + 128, 0, 0);
    v16bf an  = loadA(aRow + kc + kb);
    v16bf bn0 = loadB(b0 + kc, hi);
    v16bf bn1 = loadB(b0 + 16 * DM + kc, hi);
    v16bf bn2 = loadB(b0 + 32 * DM + kc, hi);
    v16bf bn3 = loadB(b0 + 48 * DM + kc, hi);
    acc[0] = wmma_bf16(a, bt0, acc[0]);
    acc[1] = wmma_bf16(a, bt1, acc[1]);
    acc[2] = wmma_bf16(a, bt2, acc[2]);
    acc[3] = wmma_bf16(a, bt3, acc[3]);
    a = an; bt0 = bn0; bt1 = bn1; bt2 = bn2; bt3 = bn3;
  }
  acc[0] = wmma_bf16(a, bt0, acc[0]);
  acc[1] = wmma_bf16(a, bt1, acc[1]);
  acc[2] = wmma_bf16(a, bt2, acc[2]);
  acc[3] = wmma_bf16(a, bt3, acc[3]);

#pragma unroll
  for (int t = 0; t < 4; ++t) {
    const float bv = bias[nt * 64 + t * 16 + r];
#pragma unroll
    for (int i = 0; i < 8; ++i) {
      const int row = mt * 16 + i + 8 * hi;
      out[(size_t)row * DM + nt * 64 + t * 16 + r] = acc[t][i] + bv;
    }
  }
}

// ---------------------------------------------------------------------------
// host launcher
// ---------------------------------------------------------------------------
extern "C" void kernel_launch(void* const* d_in, const int* in_sizes, int n_in,
                              void* d_out, int out_size, void* d_ws,
                              size_t ws_size, hipStream_t stream) {
  (void)in_sizes; (void)n_in; (void)out_size; (void)ws_size;

  const float* q      = (const float*)d_in[0];
  const unsigned char* mask = (const unsigned char*)d_in[1];
  const float* w_qs   = (const float*)d_in[2];
  const float* w_ks   = (const float*)d_in[3];
  const float* w_vs   = (const float*)d_in[4];
  const float* proj_w = (const float*)d_in[5];
  const float* proj_b = (const float*)d_in[6];
  float* out = (float*)d_out;

  unsigned char* ws = (unsigned char*)d_ws;
  const size_t SZ_QBF  = (size_t)MTOT * DM * 2;
  const size_t SZ_WT   = (size_t)NH * DK * DM * 2;
  const size_t SZ_PW   = (size_t)DM * DM * 2;
  const size_t SZ_HEAD = (size_t)NH * MTOT * DK * 2;
  size_t off = 0;
  auto take = [&](size_t bytes) { size_t o = off; off += (bytes + 255) & ~(size_t)255; return o; };
  unsigned short* qbf  = (unsigned short*)(ws + take(SZ_QBF));
  unsigned short* wqT  = (unsigned short*)(ws + take(SZ_WT));
  unsigned short* wkT  = (unsigned short*)(ws + take(SZ_WT));
  unsigned short* wvT  = (unsigned short*)(ws + take(SZ_WT));
  unsigned short* pwbf = (unsigned short*)(ws + take(SZ_PW));
  unsigned short* Qh   = (unsigned short*)(ws + take(SZ_HEAD));
  unsigned short* Kh   = (unsigned short*)(ws + take(SZ_HEAD));
  unsigned short* VTb  = (unsigned short*)(ws + take(SZ_HEAD));
  unsigned short* attn = (unsigned short*)(ws + take(SZ_QBF));

  k_cvt<<<4096, 256, 0, stream>>>(q, qbf, MTOT * DM);
  k_cvt_wT<<<2048, 256, 0, stream>>>(w_qs, wqT);
  k_cvt_wT<<<2048, 256, 0, stream>>>(w_ks, wkT);
  k_cvt_wT<<<2048, 256, 0, stream>>>(w_vs, wvT);
  k_cvt<<<2048, 256, 0, stream>>>(proj_w, pwbf, DM * DM);

  qkv_gemm<<<dim3(128, NH), 128, 0, stream>>>(qbf, wqT, Qh, 0);
  qkv_gemm<<<dim3(128, NH), 128, 0, stream>>>(qbf, wkT, Kh, 0);
  qkv_gemm<<<dim3(128, NH), 128, 0, stream>>>(qbf, wvT, VTb, 1);

  attn_fwd<<<dim3(SS / 16 / 4, NH, BB), 128, 0, stream>>>(Qh, Kh, VTb, mask, attn);

  out_proj<<<dim3(128, 16), 128, 0, stream>>>(attn, pwbf, proj_b, out);
}
